// CharsLstm_9277129359705
// MI455X (gfx1250) — matrix operand order repaired and checked by reading, not compiled, not run on hardware
//
#include <hip/hip_runtime.h>
#include <hip/hip_bf16.h>
#include <stdint.h>

// Problem dims (fixed by reference)
#define B_WORDS 4096
#define T_STEPS 30
#define D_IN    512
#define H_DIM   1024
#define V_CHARS 128
#define G4H     4096   // 4*H

typedef __attribute__((ext_vector_type(16))) __bf16 v16bf;
typedef __attribute__((ext_vector_type(8)))  float  v8f;

__device__ __forceinline__ unsigned short f2bf(float f) {
  unsigned u = __float_as_uint(f);
  u += 0x7FFFu + ((u >> 16) & 1u);   // round-to-nearest-even
  return (unsigned short)(u >> 16);
}

// CDNA5 async global->LDS copy (ASYNCcnt-tracked, no VGPR round trip).
// Generic LDS pointers truncate to the raw LDS byte address (ISA 10.2 aperture rule).
__device__ __forceinline__ void async_copy_b128(void* lds_ptr, const void* gptr) {
  unsigned lds = (unsigned)(size_t)lds_ptr;
  asm volatile("global_load_async_to_lds_b128 %0, %1, off"
               :: "v"(lds), "v"(gptr) : "memory");
}
__device__ __forceinline__ void wait_async0() {
  asm volatile("s_wait_asynccnt 0" ::: "memory");
}

// ---------------------------------------------------------------- fp32 -> bf16
__global__ __launch_bounds__(256) void k_f32_to_bf16(const float* __restrict__ in,
                                                     unsigned short* __restrict__ out,
                                                     int n) {
  int i = blockIdx.x * 256 + threadIdx.x;
  if (i < n) out[i] = f2bf(in[i]);
}

// --------------------------------------------- Gx[v][n] = (emb @ W_ih.T)[v][n] + b_ih[n] + b_hh[n]
// grid: (16 n-chunks of 256, 16 v-chunks of 8), block 256
__global__ __launch_bounds__(256) void k_gx(const float* __restrict__ emb,
                                            const float* __restrict__ W_ih,
                                            const float* __restrict__ b_ih,
                                            const float* __restrict__ b_hh,
                                            float* __restrict__ Gx) {
  __shared__ float se[8][D_IN];
  const int tx = threadIdx.x;
  const int n  = blockIdx.x * 256 + tx;   // 0..4095
  const int v0 = blockIdx.y * 8;          // 0..120
  #pragma unroll
  for (int i = 0; i < 16; ++i) {
    int e = tx + i * 256;                 // 0..4095 = 8 rows * 512
    se[e >> 9][e & 511] = emb[(size_t)(v0 + (e >> 9)) * D_IN + (e & 511)];
  }
  __syncthreads();
  float acc[8] = {};
  const float* wr = W_ih + (size_t)n * D_IN;
  for (int d = 0; d < D_IN; ++d) {
    float w = wr[d];
    #pragma unroll
    for (int v = 0; v < 8; ++v) acc[v] += se[v][d] * w;
  }
  float bias = b_ih[n] + b_hh[n];
  #pragma unroll
  for (int v = 0; v < 8; ++v)
    Gx[(size_t)(v0 + v) * G4H + n] = acc[v] + bias;
}

// ---------------------------------------------------- recurrent GEMM (WMMA bf16)
// gates[B,4H] = h_bf16[B,H] @ W_hh_bf16[4H,H].T   (both row-major, K contiguous)
// block = 256 thr (8 waves), tile 128x128, BK=32; wave tile 64Mx32N = 4x2 WMMA accs.
// Double-buffered LDS fed by global_load_async_to_lds_b128.
__global__ __launch_bounds__(256) void k_lstm_gemm(const unsigned short* __restrict__ A,
                                                   const unsigned short* __restrict__ Bm,
                                                   float* __restrict__ C) {
  __shared__ unsigned short sA[2][128 * 32];
  __shared__ unsigned short sB[2][128 * 32];
  const int tid  = threadIdx.x;
  const int m0   = blockIdx.x * 128;
  const int n0   = blockIdx.y * 128;
  const int lane = tid & 31;
  const int wave = tid >> 5;
  const int wm   = (wave >> 2) * 64;   // 0 or 64
  const int wn   = (wave & 3) * 32;    // 0,32,64,96
  const int lrow = tid >> 2;           // 0..63
  const int lcol = (tid & 3) * 8;      // half-offset 0,8,16,24 (16B chunks)

  v8f acc[4][2] = {};

  const unsigned short* ga0 = A  + (size_t)(m0 + lrow)      * H_DIM + lcol;
  const unsigned short* ga1 = A  + (size_t)(m0 + lrow + 64) * H_DIM + lcol;
  const unsigned short* gb0 = Bm + (size_t)(n0 + lrow)      * H_DIM + lcol;
  const unsigned short* gb1 = Bm + (size_t)(n0 + lrow + 64) * H_DIM + lcol;

  auto stage = [&](int buf, int k) {
    async_copy_b128(&sA[buf][lrow * 32 + lcol],        ga0 + k);
    async_copy_b128(&sA[buf][(lrow + 64) * 32 + lcol], ga1 + k);
    async_copy_b128(&sB[buf][lrow * 32 + lcol],        gb0 + k);
    async_copy_b128(&sB[buf][(lrow + 64) * 32 + lcol], gb1 + k);
  };

  // prologue: fill buffer 0 with K-slice 0
  stage(0, 0);
  wait_async0();
  __syncthreads();

  for (int k = 0; k < H_DIM; k += 32) {
    const int cur = (k >> 5) & 1;
    if (k + 32 < H_DIM) stage(cur ^ 1, k + 32);   // overlap next slice with compute

    union Frag { v16bf v; uint4 q[2]; };
    Frag a[4], b[2];
    const int r  = lane & 15;
    const int hs = lane >> 4;
    // A fragment: lanes 0-15 hold K{0..7,16..23}, lanes 16-31 hold K{8..15,24..31}
    #pragma unroll
    for (int i = 0; i < 4; ++i) {
      const unsigned short* p = &sA[cur][(wm + i * 16 + r) * 32];
      a[i].q[0] = *(const uint4*)(p + hs * 8);
      a[i].q[1] = *(const uint4*)(p + 16 + hs * 8);
    }
    // B fragment: lane holds column n = wn + j*16 + r; lanes 0-15 K{0..7,16..23}, 16-31 K{8..15,24..31}
    #pragma unroll
    for (int j = 0; j < 2; ++j) {
      const unsigned short* p = &sB[cur][(wn + j * 16 + r) * 32];
      b[j].q[0] = *(const uint4*)(p + hs * 8);
      b[j].q[1] = *(const uint4*)(p + 16 + hs * 8);
    }
    #pragma unroll
    for (int i = 0; i < 4; ++i)
      #pragma unroll
      for (int j = 0; j < 2; ++j)
        acc[i][j] = __builtin_amdgcn_wmma_f32_16x16x32_bf16(
            false, a[i].v, false, b[j].v, (short)0, acc[i][j], false, false);

    wait_async0();     // this wave's async copies into nxt are done
    __syncthreads();   // everyone's are; safe to read nxt / overwrite cur next iter
  }

  // epilogue: C VGPR layout — VGPR r: M = r (lanes 0-15) / 8+r (lanes 16-31), N = lane&15
  const int r  = lane & 15;
  const int hs = lane >> 4;
  #pragma unroll
  for (int i = 0; i < 4; ++i) {
    #pragma unroll
    for (int j = 0; j < 2; ++j) {
      const int mb = m0 + wm + i * 16 + hs * 8;
      const int nb = n0 + wn + j * 16 + r;
      #pragma unroll
      for (int rr = 0; rr < 8; ++rr)
        C[(size_t)(mb + rr) * G4H + nb] = acc[i][j][rr];
    }
  }
}

// -------------------------------------------------------------- LSTM cell update
__global__ __launch_bounds__(256) void k_lstm_cell(const float* __restrict__ gates,
                                                   const float* __restrict__ Gx,
                                                   const int* __restrict__ x, int t,
                                                   const float* __restrict__ cin,
                                                   float* __restrict__ cout,
                                                   unsigned short* __restrict__ hbf,
                                                   float* __restrict__ hout) {
  const int idx = blockIdx.x * 256 + threadIdx.x;   // b*H + hh
  const int b   = idx >> 10;
  const int hh  = idx & (H_DIM - 1);
  const int v   = x[(size_t)b * T_STEPS + t];
  const float* gr  = gates + (size_t)b * G4H;
  const float* gxr = Gx + (size_t)v * G4H;
  float zi = gr[hh]              + gxr[hh];
  float zf = gr[H_DIM + hh]      + gxr[H_DIM + hh];
  float zg = gr[2 * H_DIM + hh]  + gxr[2 * H_DIM + hh];
  float zo = gr[3 * H_DIM + hh]  + gxr[3 * H_DIM + hh];
  float si = 1.f / (1.f + __expf(-zi));
  float sf = 1.f / (1.f + __expf(-zf));
  float so = 1.f / (1.f + __expf(-zo));
  float c  = sf * cin[idx] + si * tanhf(zg);
  float h  = so * tanhf(c);
  cout[idx] = c;
  hbf[idx]  = f2bf(h);
  if (hout) hout[idx] = h;
}

// ------------------------------------------------------------------------ launch
extern "C" void kernel_launch(void* const* d_in, const int* in_sizes, int n_in,
                              void* d_out, int out_size, void* d_ws, size_t ws_size,
                              hipStream_t stream) {
  const int*   x    = (const int*)d_in[0];
  const float* emb  = (const float*)d_in[1];
  const float* W_ih = (const float*)d_in[2];
  const float* W_hh = (const float*)d_in[3];
  const float* b_ih = (const float*)d_in[4];
  const float* b_hh = (const float*)d_in[5];
  const float* h0   = (const float*)d_in[6];
  const float* c0   = (const float*)d_in[7];

  char* ws = (char*)d_ws;
  size_t o = 0;
  unsigned short* wbf = (unsigned short*)(ws + o); o += (size_t)G4H * H_DIM * 2;      // 8 MB
  float* Gx           = (float*)(ws + o);          o += (size_t)V_CHARS * G4H * 4;    // 2 MB
  float* gates        = (float*)(ws + o);          o += (size_t)B_WORDS * G4H * 4;    // 64 MB
  unsigned short* hbf = (unsigned short*)(ws + o); o += (size_t)B_WORDS * H_DIM * 2;  // 8 MB
  float* c_ws         = (float*)(ws + o);                                             // 16 MB

  // one-time per launch: bf16 weight/state, folded input-gate table
  k_f32_to_bf16<<<(G4H * H_DIM + 255) / 256, 256, 0, stream>>>(W_hh, wbf, G4H * H_DIM);
  k_f32_to_bf16<<<(B_WORDS * H_DIM + 255) / 256, 256, 0, stream>>>(h0, hbf, B_WORDS * H_DIM);
  k_gx<<<dim3(16, 16), 256, 0, stream>>>(emb, W_ih, b_ih, b_hh, Gx);

  const float* csrc = c0;
  for (int t = 0; t < T_STEPS; ++t) {
    k_lstm_gemm<<<dim3(32, 32), 256, 0, stream>>>(hbf, wbf, gates);
    k_lstm_cell<<<(B_WORDS * H_DIM) / 256, 256, 0, stream>>>(
        gates, Gx, x, t, csrc, c_ws, hbf,
        (t == T_STEPS - 1) ? (float*)d_out : nullptr);
    csrc = c_ws;
  }
}
